// Gate_24498493456498
// MI455X (gfx1250) — compile-verified, matrix operand order checked by reading, not run on hardware
//
#include <hip/hip_runtime.h>

#define TOKENS 32768
#define DIM 2048
#define N_EXPERTS 64
#define TOP_K 6
#define ROUTE_SCALE 1.0f

typedef _Float16 v16h __attribute__((ext_vector_type(16)));
typedef _Float16 v8h  __attribute__((ext_vector_type(8)));
typedef _Float16 v4h  __attribute__((ext_vector_type(4)));
typedef float    v8f  __attribute__((ext_vector_type(8)));
typedef float    v4f  __attribute__((ext_vector_type(4)));

// ---------------------------------------------------------------------------
// Kernel 1: convert router weights f32 -> f16 into workspace (256 KB, L2-hot).
// Layout stays [expert][dim] row-major, which is exactly the B^T row order the
// WMMA B-fragment wants.
// ---------------------------------------------------------------------------
__global__ __launch_bounds__(256)
void moe_gate_wcvt(const float* __restrict__ w, _Float16* __restrict__ wh) {
    int i = (blockIdx.x * 256 + threadIdx.x) * 4;   // total = 64*2048, exact
    v4f  src = *(const v4f*)(w + i);
    v4h  dst;
    dst[0] = (_Float16)src[0];
    dst[1] = (_Float16)src[1];
    dst[2] = (_Float16)src[2];
    dst[3] = (_Float16)src[3];
    *(v4h*)(wh + i) = dst;
}

// ---------------------------------------------------------------------------
// Kernel 2: gate. 4 waves/block; wave w computes tokens [block*64 + w*16, +16)
// against all 64 experts via 4 accumulator tiles of v_wmma_f32_16x16x32_f16.
// ---------------------------------------------------------------------------
__global__ __launch_bounds__(128)
void moe_gate_main(const float* __restrict__ x,
                   const _Float16* __restrict__ wh,
                   const float* __restrict__ bias,
                   float* __restrict__ out_w,
                   int* __restrict__ out_i) {
    __shared__ float lds[64 * 65];                  // 64 tokens x 64 experts, pad->65

    const int lane = threadIdx.x & 31;
    const int wave = threadIdx.x >> 5;
    const int m    = lane & 15;                     // row (A) / col (B) inside tile
    const int sel  = (lane & 16) ? 8 : 0;           // K sub-select per ISA layout
    const int tokenBase = blockIdx.x * 64 + wave * 16;

    const float*    rowA  = x  + (size_t)(tokenBase + m) * DIM;
    const _Float16* rowB0 = wh + (size_t)( 0 + m) * DIM;
    const _Float16* rowB1 = wh + (size_t)(16 + m) * DIM;
    const _Float16* rowB2 = wh + (size_t)(32 + m) * DIM;
    const _Float16* rowB3 = wh + (size_t)(48 + m) * DIM;

    v8f c0 = {}, c1 = {}, c2 = {}, c3 = {};

    for (int kc = 0; kc < DIM; kc += 32) {
        // ---- A fragment: 16 f32 -> f16 inline (K = kc+sel..+7, kc+16+sel..+7)
        v4f a0 = *(const v4f*)(rowA + kc + sel);
        v4f a1 = *(const v4f*)(rowA + kc + sel + 4);
        v4f a2 = *(const v4f*)(rowA + kc + sel + 16);
        v4f a3 = *(const v4f*)(rowA + kc + sel + 20);
        v16h a;
        a[0]  = (_Float16)a0[0]; a[1]  = (_Float16)a0[1];
        a[2]  = (_Float16)a0[2]; a[3]  = (_Float16)a0[3];
        a[4]  = (_Float16)a1[0]; a[5]  = (_Float16)a1[1];
        a[6]  = (_Float16)a1[2]; a[7]  = (_Float16)a1[3];
        a[8]  = (_Float16)a2[0]; a[9]  = (_Float16)a2[1];
        a[10] = (_Float16)a2[2]; a[11] = (_Float16)a2[3];
        a[12] = (_Float16)a3[0]; a[13] = (_Float16)a3[1];
        a[14] = (_Float16)a3[2]; a[15] = (_Float16)a3[3];

        // ---- B fragments: already f16, two 16-byte loads per tile
        v16h b0v, b1v, b2v, b3v;
        {
            v8h lo = *(const v8h*)(rowB0 + kc + sel);
            v8h hi = *(const v8h*)(rowB0 + kc + sel + 16);
            #pragma unroll
            for (int i = 0; i < 8; ++i) { b0v[i] = lo[i]; b0v[8 + i] = hi[i]; }
        }
        {
            v8h lo = *(const v8h*)(rowB1 + kc + sel);
            v8h hi = *(const v8h*)(rowB1 + kc + sel + 16);
            #pragma unroll
            for (int i = 0; i < 8; ++i) { b1v[i] = lo[i]; b1v[8 + i] = hi[i]; }
        }
        {
            v8h lo = *(const v8h*)(rowB2 + kc + sel);
            v8h hi = *(const v8h*)(rowB2 + kc + sel + 16);
            #pragma unroll
            for (int i = 0; i < 8; ++i) { b2v[i] = lo[i]; b2v[8 + i] = hi[i]; }
        }
        {
            v8h lo = *(const v8h*)(rowB3 + kc + sel);
            v8h hi = *(const v8h*)(rowB3 + kc + sel + 16);
            #pragma unroll
            for (int i = 0; i < 8; ++i) { b3v[i] = lo[i]; b3v[8 + i] = hi[i]; }
        }

        // ---- D = A*B + C, f32 accumulate (one A fragment reused 4x)
        c0 = __builtin_amdgcn_wmma_f32_16x16x32_f16(false, a, false, b0v, (short)0, c0, false, false);
        c1 = __builtin_amdgcn_wmma_f32_16x16x32_f16(false, a, false, b1v, (short)0, c1, false, false);
        c2 = __builtin_amdgcn_wmma_f32_16x16x32_f16(false, a, false, b2v, (short)0, c2, false, false);
        c3 = __builtin_amdgcn_wmma_f32_16x16x32_f16(false, a, false, b3v, (short)0, c3, false, false);
    }

    // ---- Scatter accumulators to LDS. C/D layout: VGPR v of lanes 0-15 holds
    // M=v, lanes 16-31 hold M=v+8; N = lane%16.
    {
        const int mrow = wave * 16 + ((lane & 16) ? 8 : 0);
        const int e    = lane & 15;
        #pragma unroll
        for (int v = 0; v < 8; ++v) {
            lds[(mrow + v) * 65 + e +  0] = c0[v];
            lds[(mrow + v) * 65 + e + 16] = c1[v];
            lds[(mrow + v) * 65 + e + 32] = c2[v];
            lds[(mrow + v) * 65 + e + 48] = c3[v];
        }
    }
    __syncthreads();

    // ---- Phase 2: one thread per token: stable softmax + top-6 on score+bias
    const int t = threadIdx.x;
    if (t < 64) {
        const float* row = &lds[t * 65];

        float mx = -3.4e38f;
        #pragma unroll 8
        for (int e = 0; e < N_EXPERTS; ++e) mx = fmaxf(mx, row[e]);

        float Z = 0.0f;
        #pragma unroll 8
        for (int e = 0; e < N_EXPERTS; ++e) Z += __expf(row[e] - mx);
        const float invZ = 1.0f / Z;

        float bv[TOP_K];
        int   bi[TOP_K];
        #pragma unroll
        for (int k = 0; k < TOP_K; ++k) { bv[k] = -3.4e38f; bi[k] = 0; }

        for (int e = 0; e < N_EXPERTS; ++e) {
            const float score  = __expf(row[e] - mx) * invZ;
            const float biased = score + bias[e];
            if (biased > bv[TOP_K - 1]) {           // strict > : lower index wins ties
                int k = TOP_K - 1;
                #pragma unroll
                for (int j = TOP_K - 1; j > 0; --j) {
                    if (biased > bv[j - 1]) {
                        bv[j] = bv[j - 1]; bi[j] = bi[j - 1]; k = j - 1;
                    }
                }
                bv[k] = biased; bi[k] = e;
            }
        }

        const size_t tok = (size_t)blockIdx.x * 64 + t;
        #pragma unroll
        for (int k = 0; k < TOP_K; ++k) {
            const float score = __expf(row[bi[k]] - mx) * invZ;
            out_w[tok * TOP_K + k] = score * ROUTE_SCALE;   // original (unbiased) score
            out_i[tok * TOP_K + k] = bi[k];
        }
    }
}

// ---------------------------------------------------------------------------
extern "C" void kernel_launch(void* const* d_in, const int* in_sizes, int n_in,
                              void* d_out, int out_size, void* d_ws, size_t ws_size,
                              hipStream_t stream) {
    const float* x    = (const float*)d_in[0];   // [TOKENS, DIM] f32
    const float* w    = (const float*)d_in[1];   // [N_EXPERTS, DIM] f32
    const float* bias = (const float*)d_in[2];   // [N_EXPERTS] f32

    _Float16* wh = (_Float16*)d_ws;              // 64*2048*2 = 256 KB scratch

    float* out_w = (float*)d_out;                                  // [TOKENS, TOP_K] f32
    int*   out_i = (int*)((float*)d_out + (size_t)TOKENS * TOP_K); // [TOKENS, TOP_K] i32

    // weight f32 -> f16 (131072 elems, 4/thread)
    moe_gate_wcvt<<<(N_EXPERTS * DIM) / (256 * 4), 256, 0, stream>>>(w, wh);

    // main gate: 64 tokens per block, 4 waves of 16 tokens x 64 experts
    moe_gate_main<<<TOKENS / 64, 128, 0, stream>>>(x, wh, bias, out_w, out_i);
}
